// MobileNet100_5145370821262
// MI455X (gfx1250) — matrix operand order repaired and checked by reading, not compiled, not run on hardware
//
#include <hip/hip_runtime.h>
#include <hip/hip_bf16.h>
#include <float.h>

typedef __attribute__((ext_vector_type(16))) _Float16 v16h;
typedef __attribute__((ext_vector_type(8)))  float    v8f;

#define B_BATCH 64
#define HW      784   // 28*28
#define CCH     256
#define NPARTS  2
#define DDIM    128
#define KCODE   1024
#define MROWS   (B_BATCH * HW * NPARTS)   // 100352

// ---------------------------------------------------------------------------
// Kernel 1: conv3x3 stride 8 pad 1 + ReLU, output NHWC ze[b*784+n][256]
// One block per (b, oy). Weights (27.6KB) + 3x3x224 input strip (8KB) in LDS.
// ---------------------------------------------------------------------------
__global__ __launch_bounds__(256) void conv_relu_kernel(
    const float* __restrict__ X,    // [64,3,224,224]
    const float* __restrict__ We,   // [256,3,3,3]
    float* __restrict__ ze)         // [64*784, 256]
{
    __shared__ float w[CCH * 27];          // 27648 B
    __shared__ float strip[3 * 3 * 224];   // 8064 B  (ci, r, col)
    const int b  = blockIdx.x;
    const int oy = blockIdx.y;
    const int t  = threadIdx.x;

    for (int i = t; i < CCH * 27; i += 256) w[i] = We[i];
    for (int i = t; i < 3 * 3 * 224; i += 256) {
        int ci  = i / (3 * 224);
        int rr  = (i / 224) % 3;
        int col = i % 224;
        int iy  = oy * 8 - 1 + rr;         // pad=1; iy in [-1, 217]
        float v = 0.0f;
        if (iy >= 0) v = X[(((size_t)b * 3 + ci) * 224 + iy) * 224 + col];
        strip[i] = v;
    }
    __syncthreads();

    const int c = t;   // channel
    for (int ox = 0; ox < 28; ++ox) {
        float acc = 0.0f;
        #pragma unroll
        for (int ci = 0; ci < 3; ++ci)
            #pragma unroll
            for (int r = 0; r < 3; ++r)
                #pragma unroll
                for (int s = 0; s < 3; ++s) {
                    int ix = ox * 8 - 1 + s;           // [-1, 217]
                    float pv = (ix >= 0) ? strip[(ci * 3 + r) * 224 + ix] : 0.0f;
                    acc = fmaf(pv, w[c * 27 + ci * 9 + r * 3 + s], acc);
                }
        ze[((size_t)b * HW + oy * 28 + ox) * CCH + c] = fmaxf(acc, 0.0f);
    }
}

// ---------------------------------------------------------------------------
// Kernel 2: codebook -> f16 copy + row norms ||c_k||^2
// ---------------------------------------------------------------------------
__global__ __launch_bounds__(128) void cbprep_kernel(
    const float* __restrict__ cb,   // [1024,128]
    _Float16* __restrict__ cbh,     // [1024,128]
    float* __restrict__ cnorm)      // [1024]
{
    const int k = blockIdx.x;
    const int d = threadIdx.x;
    float v = cb[(size_t)k * DDIM + d];
    cbh[(size_t)k * DDIM + d] = (_Float16)v;
    __shared__ float s[128];
    s[d] = v * v;
    __syncthreads();
    for (int off = 64; off > 0; off >>= 1) {
        if (d < off) s[d] += s[d + off];
        __syncthreads();
    }
    if (d == 0) cnorm[k] = s[0];
}

// ---------------------------------------------------------------------------
// Kernel 3: fused distance GEMM + argmin via v_wmma_f32_16x16x32_f16.
// One wave owns a 16-row M tile (A fragments held in registers for all 64
// N tiles).  B tiles (16 codebook rows = 4KB f16) are double-buffered in LDS,
// staged with GLOBAL_LOAD_ASYNC_TO_LDS_B128 (ASYNCcnt) and shared by all 8
// waves of the block -> 8x less L2 traffic than per-wave global loads.
// score = ||c||^2 - 2*dot (the ||z||^2 term does not affect the argmin).
// C/D layout: lane l holds N = l%16, rows M = r + 8*(l/16), r=0..7.
// A 16x32 f16 order used here (same k order as B): g=l/16, M=l%16, pair v: K = 2*v+16g
// B 32x16 f16 order used here (same k order as A): lane l holds N=l%16, halves j=0..15 -> K = j + 16*g.
// ---------------------------------------------------------------------------
__global__ __launch_bounds__(256) void vq_argmin_kernel(
    const float* __restrict__ ze,       // [MROWS,128]
    const _Float16* __restrict__ cbh,   // [1024,128]
    const float* __restrict__ cnorm,    // [1024]
    int* __restrict__ idx)              // [MROWS]
{
    __shared__ __align__(32) _Float16 bbuf[2][16 * DDIM];   // 2 x 4KB

    const int lane = threadIdx.x & 31;
    const int wave = threadIdx.x >> 5;
    const int mtile = blockIdx.x * 8 + wave;
    const int m0 = mtile * 16;
    const int g  = lane >> 4;
    const int lm = lane & 15;

    // async stage of one 16x128-half B tile: 256 threads x 16B = 4KB
    auto stage = [&](int nt, int buf) {
        const int i   = threadIdx.x;
        const int row = i >> 4;          // 0..15 codebook row within tile
        const int ch  = i & 15;          // 16-byte chunk within row
        const _Float16* gp = cbh + ((size_t)(nt * 16 + row) * DDIM + ch * 8);
        unsigned ldsoff = (unsigned)(uintptr_t)&bbuf[buf][row * DDIM + ch * 8];
        asm volatile("global_load_async_to_lds_b128 %0, %1, off"
                     :: "v"(ldsoff), "v"(gp) : "memory");
    };

    // ---- load + convert A fragments (reused across all 64 N tiles)
    v16h a[4];
    const float* zrow = ze + (size_t)(m0 + lm) * DDIM;
    #pragma unroll
    for (int c = 0; c < 4; ++c) {
        #pragma unroll
        for (int v = 0; v < 8; ++v) {
            int kbase = 32 * c + 2 * v + 16 * g;
            a[c][2 * v]     = (_Float16)zrow[kbase];
            a[c][2 * v + 1] = (_Float16)zrow[kbase + 1];
        }
    }

    float mv[8];
    int   mi[8];
    #pragma unroll
    for (int r = 0; r < 8; ++r) { mv[r] = FLT_MAX; mi[r] = 0; }

    stage(0, 0);                              // prologue: tile 0 -> buffer 0

    for (int nt = 0; nt < 64; ++nt) {
        const int cur = nt & 1;
        if (nt + 1 < 64) {
            stage(nt + 1, cur ^ 1);           // overlap next stage w/ compute
            asm volatile("s_wait_asynccnt 0x1" ::: "memory"); // tile nt done
        } else {
            asm volatile("s_wait_asynccnt 0x0" ::: "memory");
        }
        __syncthreads();                      // tile nt visible to all waves

        const int n = nt * 16 + lm;
        v8f acc = {};
        #pragma unroll
        for (int c = 0; c < 4; ++c) {
            v16h bfrag = *(const v16h*)&bbuf[cur][lm * DDIM + 32 * c + 16 * g];
            acc = __builtin_amdgcn_wmma_f32_16x16x32_f16(
                false, a[c], false, bfrag, (short)0, acc, false, false);
        }
        const float cn = cnorm[n];
        #pragma unroll
        for (int r = 0; r < 8; ++r) {
            float s = fmaf(-2.0f, acc[r], cn);
            if (s < mv[r]) { mv[r] = s; mi[r] = n; }
        }
        __syncthreads();                      // all waves done with buf[cur]
    }

    // reduce over the 16 lanes (same half-wave) that hold different N
    #pragma unroll
    for (int off = 1; off < 16; off <<= 1) {
        #pragma unroll
        for (int r = 0; r < 8; ++r) {
            float ov = __shfl_xor(mv[r], off, 32);
            int   oi = __shfl_xor(mi[r], off, 32);
            if (ov < mv[r] || (ov == mv[r] && oi < mi[r])) { mv[r] = ov; mi[r] = oi; }
        }
    }
    if (lm == 0) {
        #pragma unroll
        for (int r = 0; r < 8; ++r) idx[m0 + r + 8 * g] = mi[r];
    }
}

// ---------------------------------------------------------------------------
// Kernel 4: gather codebook rows, accumulate pooled[b][c] and commit partials
// One block per batch element b; thread t = p*128 + dd.
// ---------------------------------------------------------------------------
__global__ __launch_bounds__(256) void gather_pool_kernel(
    const float* __restrict__ ze,    // [B*HW, 256]
    const float* __restrict__ cb,    // [1024,128] fp32
    const int* __restrict__ idx,     // [B*HW, 2]
    float* __restrict__ pooled,      // [64,256]
    float* __restrict__ cpart)       // [64]
{
    const int b = blockIdx.x;
    const int t = threadIdx.x;
    const int p  = t >> 7;
    const int dd = t & 127;
    float pacc = 0.0f, cacc = 0.0f;
    for (int n = 0; n < HW; ++n) {
        const int row = b * HW + n;
        const int k = idx[row * 2 + p];
        const float q  = cb[(size_t)k * DDIM + dd];
        const float zv = ze[(size_t)row * CCH + t];
        pacc += q;
        const float df = q - zv;
        cacc = fmaf(df, df, cacc);
    }
    pooled[b * CCH + t] = pacc * (1.0f / (float)HW);
    __shared__ float s[256];
    s[t] = cacc;
    __syncthreads();
    for (int off = 128; off > 0; off >>= 1) {
        if (t < off) s[t] += s[t + off];
        __syncthreads();
    }
    if (t == 0) cpart[b] = s[0];
}

// ---------------------------------------------------------------------------
// Kernel 5: head GEMM logits = pooled @ Wd^T  (64x1000x256) + commit scalar
// Blocks 0..63: one batch row each. Block 64: reduce commit loss.
// ---------------------------------------------------------------------------
__global__ __launch_bounds__(256) void head_kernel(
    const float* __restrict__ pooled,   // [64,256]
    const float* __restrict__ Wd,       // [1000,256]
    const float* __restrict__ cpart,    // [64]
    float* __restrict__ out)            // [64*1000 logits, +1 commit]
{
    const int t = threadIdx.x;
    if (blockIdx.x == B_BATCH) {
        __shared__ float s[64];
        if (t < 64) s[t] = cpart[t];
        __syncthreads();
        if (t == 0) {
            float tot = 0.0f;
            for (int i = 0; i < 64; ++i) tot += s[i];
            // commit = P * COMMITMENT * mean over [B,HW,P,d]
            out[B_BATCH * 1000] =
                2.0f * tot / ((float)B_BATCH * HW * NPARTS * DDIM);
        }
        return;
    }
    const int b = blockIdx.x;
    __shared__ float pb[CCH];
    pb[t] = pooled[b * CCH + t];
    __syncthreads();
    for (int o = t; o < 1000; o += 256) {
        const float* wr = Wd + (size_t)o * CCH;
        float acc = 0.0f;
        #pragma unroll 8
        for (int c = 0; c < CCH; ++c) acc = fmaf(pb[c], wr[c], acc);
        out[b * 1000 + o] = acc;
    }
}

// ---------------------------------------------------------------------------
extern "C" void kernel_launch(void* const* d_in, const int* in_sizes, int n_in,
                              void* d_out, int out_size, void* d_ws, size_t ws_size,
                              hipStream_t stream) {
    const float* X  = (const float*)d_in[0];   // [64,3,224,224]
    const float* We = (const float*)d_in[1];   // [256,3,3,3]
    const float* Wd = (const float*)d_in[2];   // [1000,256]
    const float* cb = (const float*)d_in[3];   // [1024,128]
    float* out = (float*)d_out;                // 64*1000 + 1

    char* ws = (char*)d_ws;
    size_t off = 0;
    auto carve = [&](size_t bytes) -> void* {
        void* p = ws + off;
        off += (bytes + 255) & ~(size_t)255;
        return p;
    };
    float*     ze     = (float*)    carve((size_t)B_BATCH * HW * CCH * 4); // 51.4 MB
    _Float16*  cbh    = (_Float16*) carve((size_t)KCODE * DDIM * 2);
    float*     cnorm  = (float*)    carve((size_t)KCODE * 4);
    int*       idx    = (int*)      carve((size_t)MROWS * 4);
    float*     pooled = (float*)    carve((size_t)B_BATCH * CCH * 4);
    float*     cpart  = (float*)    carve((size_t)B_BATCH * 4);

    dim3 convGrid(B_BATCH, 28);
    conv_relu_kernel<<<convGrid, 256, 0, stream>>>(X, We, ze);
    cbprep_kernel<<<KCODE, 128, 0, stream>>>(cb, cbh, cnorm);
    vq_argmin_kernel<<<MROWS / (16 * 8), 256, 0, stream>>>(ze, cbh, cnorm, idx);
    gather_pool_kernel<<<B_BATCH, 256, 0, stream>>>(ze, cb, idx, pooled, cpart);
    head_kernel<<<B_BATCH + 1, 256, 0, stream>>>(pooled, Wd, cpart, out);
}
